// GATLayer_18751827214956
// MI455X (gfx1250) — matrix operand order repaired and checked by reading, not compiled, not run on hardware
//
#include <hip/hip_runtime.h>

#define B_ 2
#define N_ 2048
#define D_ 512
#define ALPHA_ 0.2f
#define EPS_ 1e-6f

typedef __bf16 v16bf __attribute__((ext_vector_type(16)));
typedef float v8f __attribute__((ext_vector_type(8)));
typedef unsigned int v8u __attribute__((ext_vector_type(8)));
typedef int v4i_t __attribute__((ext_vector_type(4)));
typedef __attribute__((address_space(3))) v4i_t* lds_v4i_p;

#if __has_builtin(__builtin_amdgcn_global_load_async_to_lds_b128) && \
    __has_builtin(__builtin_amdgcn_s_wait_asynccnt)
#define ASYNC_LDS 1
#else
#define ASYNC_LDS 0
#endif

__device__ __forceinline__ void async_cp16(void* lds, const void* g) {
#if ASYNC_LDS
    __builtin_amdgcn_global_load_async_to_lds_b128((v4i_t*)g, (lds_v4i_p)lds, 0, 0);
#else
    *(uint4*)lds = *(const uint4*)g;
#endif
}
__device__ __forceinline__ void async_wait() {
#if ASYNC_LDS
    __builtin_amdgcn_s_wait_asynccnt(0);
#endif
}

__device__ __forceinline__ unsigned short f2bf(float f) {
    unsigned int u = __builtin_bit_cast(unsigned int, f);
    u += 0x7fffu + ((u >> 16) & 1u);   // round-to-nearest-even
    return (unsigned short)(u >> 16);
}
__device__ __forceinline__ float bf2f(unsigned short h) {
    unsigned int u = ((unsigned int)h) << 16;
    return __builtin_bit_cast(float, u);
}
__device__ __forceinline__ float wave_sum(float v) {
    #pragma unroll
    for (int m = 16; m >= 1; m >>= 1) v += __shfl_xor(v, m, 32);
    return v;
}
__device__ __forceinline__ float wave_max(float v) {
    #pragma unroll
    for (int m = 16; m >= 1; m >>= 1) v = fmaxf(v, __shfl_xor(v, m, 32));
    return v;
}

// ---------------- LayerNorm -> bf16 (one wave32 per row of D=512) ----------------
__global__ void __launch_bounds__(256) ln_bf16_kernel(const float* __restrict__ x,
                                                      const float* __restrict__ g,
                                                      const float* __restrict__ bta,
                                                      unsigned short* __restrict__ out) {
    int row  = blockIdx.x * 8 + (threadIdx.x >> 5);
    int lane = threadIdx.x & 31;
    const float* xr = x + (size_t)row * D_;
    float vals[16];
    float s = 0.f;
    #pragma unroll
    for (int i = 0; i < 16; i++) { vals[i] = xr[lane + i * 32]; s += vals[i]; }
    float mu = wave_sum(s) * (1.0f / D_);
    float v2 = 0.f;
    #pragma unroll
    for (int i = 0; i < 16; i++) { float d = vals[i] - mu; v2 += d * d; }
    float var = wave_sum(v2) * (1.0f / D_);
    float rs = rsqrtf(var + EPS_);
    unsigned short* orow = out + (size_t)row * D_;
    #pragma unroll
    for (int i = 0; i < 16; i++) {
        int c = lane + i * 32;
        orow[c] = f2bf((vals[i] - mu) * rs * g[c] + bta[c]);
    }
}

// ---------------- W [k][o] fp32 -> Wt [o][k] bf16 ----------------
__global__ void wtrans_kernel(const float* __restrict__ W, unsigned short* __restrict__ Wt) {
    int idx = blockIdx.x * 256 + threadIdx.x;   // o*D + k
    int o = idx / D_, k = idx % D_;
    Wt[idx] = f2bf(W[(size_t)k * D_ + o]);
}

// ---------------- GEMM: h_t[b][o][n] = sum_k A[b][n][k] * Wt[o][k] ----------------
// 8 waves/block, block tile 128(M=n) x 64(O), async LDS staging, 2x2 wave tiling.
__global__ void __launch_bounds__(256) gemm_kernel(const unsigned short* __restrict__ Abf,
                                                   const unsigned short* __restrict__ Wt,
                                                   unsigned short* __restrict__ Ct) {
    __shared__ unsigned int As[128][16];   // 128 rows x 32 bf16 (as 16 u32)
    __shared__ unsigned int Bs[64][16];    // 64 o-rows x 32 bf16

    int blk = blockIdx.x;
    int ob = blk % (D_ / 64);
    int mb = (blk / (D_ / 64)) % (N_ / 128);
    int b  = blk / ((D_ / 64) * (N_ / 128));

    int tid = threadIdx.x;
    int wave = tid >> 5, lane = tid & 31;
    int wm = wave >> 1, wn = wave & 1;       // waves: 4 along M, 2 along O
    int half = lane >> 4, l16 = lane & 15;

    const unsigned int* gA = (const unsigned int*)Abf + (size_t)b * N_ * (D_ / 2);
    const unsigned int* gB = (const unsigned int*)Wt;

    // staging chunk assignment (16B per chunk)
    int ra0 = tid >> 2, ra1 = (tid + 256) >> 2;       // A rows for chunks tid, tid+256
    int qa  = tid & 3;
    int rb  = tid >> 2;                                // B rows 0..63

    v8f acc[2][2] = {};
    for (int ks = 0; ks < D_ / 32; ks++) {
        __syncthreads();   // previous iteration's fragment reads complete
        async_cp16(&As[ra0][qa * 4],
                   gA + (size_t)(mb * 128 + ra0) * (D_ / 2) + ks * 16 + qa * 4);
        async_cp16(&As[ra1][qa * 4],
                   gA + (size_t)(mb * 128 + ra1) * (D_ / 2) + ks * 16 + qa * 4);
        async_cp16(&Bs[rb & 63][qa * 4],
                   gB + (size_t)(ob * 64 + (rb & 63)) * (D_ / 2) + ks * 16 + qa * 4);
        async_wait();
        __syncthreads();

        v8u af[2], bf[2];
        #pragma unroll
        for (int mi = 0; mi < 2; mi++) {
            int m = wm * 32 + mi * 16 + l16;
            #pragma unroll
            for (int v = 0; v < 8; v++) {
                int kk = (v >= 4 ? 8 : 0) + half * 4 + (v & 3);  // A-frag u32 index
                af[mi][v] = As[m][kk];
            }
        }
        #pragma unroll
        for (int oi = 0; oi < 2; oi++) {
            int o = wn * 32 + oi * 16 + l16;
            #pragma unroll
            for (int v = 0; v < 8; v++) {
                int kk = (v >= 4 ? 8 : 0) + half * 4 + (v & 3);  // B-frag u32 index
                bf[oi][v] = Bs[o][kk];
            }
        }
        #pragma unroll
        for (int mi = 0; mi < 2; mi++)
            #pragma unroll
            for (int oi = 0; oi < 2; oi++)
                acc[mi][oi] = __builtin_amdgcn_wmma_f32_16x16x32_bf16(
                    false, __builtin_bit_cast(v16bf, af[mi]),
                    false, __builtin_bit_cast(v16bf, bf[oi]),
                    (short)0, acc[mi][oi], false, false);
    }

    // Epilogue: store transposed bf16 (Ct[b][o][n]) with packed 16B stores along n.
    unsigned short* ct = Ct + (size_t)b * D_ * N_;
    #pragma unroll
    for (int mi = 0; mi < 2; mi++) {
        #pragma unroll
        for (int oi = 0; oi < 2; oi++) {
            int o = ob * 64 + wn * 32 + oi * 16 + l16;
            int nbase = mb * 128 + wm * 32 + mi * 16 + half * 8;
            uint4 pk;
            pk.x = (unsigned int)f2bf(acc[mi][oi][0]) | ((unsigned int)f2bf(acc[mi][oi][1]) << 16);
            pk.y = (unsigned int)f2bf(acc[mi][oi][2]) | ((unsigned int)f2bf(acc[mi][oi][3]) << 16);
            pk.z = (unsigned int)f2bf(acc[mi][oi][4]) | ((unsigned int)f2bf(acc[mi][oi][5]) << 16);
            pk.w = (unsigned int)f2bf(acc[mi][oi][6]) | ((unsigned int)f2bf(acc[mi][oi][7]) << 16);
            *(uint4*)(ct + (size_t)o * N_ + nbase) = pk;
        }
    }
}

// ---------------- s/t projections from transposed h ----------------
__global__ void st_kernel(const unsigned short* __restrict__ hbt,
                          const float* __restrict__ a_src, const float* __restrict__ a_dst,
                          float* __restrict__ s, float* __restrict__ t, int H, int Dh) {
    int idx = blockIdx.x * 256 + threadIdx.x;   // b*H*N + h*N + n
    int n = idx % N_;
    int h = (idx / N_) % H;
    int b = idx / (N_ * H);
    const unsigned short* col = hbt + ((size_t)b * D_ + h * Dh) * N_ + n;
    float ss = 0.f, tt = 0.f;
    for (int k = 0; k < Dh; k++) {
        float v = bf2f(col[(size_t)k * N_]);
        ss += v * a_src[h * Dh + k];
        tt += v * a_dst[h * Dh + k];
    }
    s[idx] = ss;
    t[idx] = tt;
}

// ---------------- softmax row stats (rank-1 logits, one wave per row) ----------------
__global__ void __launch_bounds__(256) stats_kernel(const float* __restrict__ s,
                                                    const float* __restrict__ t,
                                                    const float* __restrict__ bias,
                                                    float* __restrict__ rm, float* __restrict__ rl,
                                                    int H) {
    int row  = blockIdx.x * 8 + (threadIdx.x >> 5);  // b*H*N + h*N + i
    int lane = threadIdx.x & 31;
    int i = row % N_;
    int h = (row / N_) % H;
    int b = row / (N_ * H);
    float si = s[row];
    const float* tb = t + ((size_t)b * H + h) * N_;
    const float* br = bias + ((size_t)b * N_ + i) * N_;
    float mx = -3.4e38f;
    for (int j = lane; j < N_; j += 32) {
        float e = si + tb[j];
        e = e > 0.f ? e : ALPHA_ * e;
        e += br[j];
        mx = fmaxf(mx, e);
    }
    mx = wave_max(mx);
    float sum = 0.f;
    for (int j = lane; j < N_; j += 32) {
        float e = si + tb[j];
        e = e > 0.f ? e : ALPHA_ * e;
        e += br[j];
        sum += __expf(e - mx);
    }
    sum = wave_sum(sum);
    if (lane == 0) { rm[row] = mx; rl[row] = 1.0f / sum; }
}

// ---------------- fused P*V ----------------
// Block = (b, head, 128-row i-block, 64-wide Dh slice); 8 waves each own a 16-row
// i-tile and share the 32x64 V tile staged in LDS (async) per j-step.
__global__ void __launch_bounds__(256) pv_kernel(const float* __restrict__ s,
                                                 const float* __restrict__ t,
                                                 const float* __restrict__ bias,
                                                 const float* __restrict__ rm,
                                                 const float* __restrict__ rl,
                                                 const unsigned short* __restrict__ hbt,
                                                 const float* __restrict__ res,
                                                 float* __restrict__ out,
                                                 int H, int Dh, int NSB) {
    __shared__ unsigned int Vs[64][16];   // 64 dh-rows x 32 bf16 (as 16 u32)

    int blk = blockIdx.x;
    int sl  = blk % NSB;
    int ib  = (blk / NSB) % (N_ / 128);
    int h   = (blk / (NSB * (N_ / 128))) % H;
    int b   = blk / (NSB * (N_ / 128) * H);

    int wave = threadIdx.x >> 5, lane = threadIdx.x & 31;
    int half = lane >> 4, l16 = lane & 15;
    int it = ib * 8 + wave;               // this wave's 16-row i-tile
    int i = it * 16 + l16;

    int rowid = ((b * H) + h) * N_ + i;
    float si = s[rowid];
    float mi = rm[rowid];
    float li = rl[rowid];
    const float* tb = t + ((size_t)b * H + h) * N_;
    const float* br = bias + ((size_t)b * N_ + i) * N_;

    // staging: 64 rows x 4 chunks of 16B -> one chunk per thread
    int vr = threadIdx.x >> 2, vq = threadIdx.x & 3;
    const unsigned int* vsrc =
        (const unsigned int*)(hbt + ((size_t)b * D_ + h * Dh + sl * 64 + vr) * N_) + vq * 4;

    v8f acc[4] = {};
    for (int j = 0; j < N_; j += 32) {
        __syncthreads();                        // previous fragment reads complete
        async_cp16(&Vs[vr][vq * 4], vsrc + (j >> 1));

        // Build P A-fragment (16x32) while the async copy is in flight:
        // p = exp(leaky(s_i + t_j) + bias - m_i) / l_i
        v8u pf;
        #pragma unroll
        for (int v = 0; v < 8; v++) {
            int kb = (v >= 4 ? 16 : 0) + half * 8 + (v & 3) * 2;
            float e0 = si + tb[j + kb];
            e0 = e0 > 0.f ? e0 : ALPHA_ * e0;
            e0 += br[j + kb];
            float e1 = si + tb[j + kb + 1];
            e1 = e1 > 0.f ? e1 : ALPHA_ * e1;
            e1 += br[j + kb + 1];
            float p0 = __expf(e0 - mi) * li;
            float p1 = __expf(e1 - mi) * li;
            pf[v] = (unsigned int)f2bf(p0) | ((unsigned int)f2bf(p1) << 16);
        }
        v16bf pbf = __builtin_bit_cast(v16bf, pf);

        async_wait();
        __syncthreads();

        #pragma unroll
        for (int oi = 0; oi < 4; oi++) {
            v8u vf;
            #pragma unroll
            for (int v = 0; v < 8; v++) vf[v] = Vs[oi * 16 + l16][(v >= 4 ? 8 : 0) + half * 4 + (v & 3)];
            acc[oi] = __builtin_amdgcn_wmma_f32_16x16x32_bf16(
                false, pbf, false, __builtin_bit_cast(v16bf, vf),
                (short)0, acc[oi], false, false);
        }
    }

    // Epilogue: out[b][i][h*Dh + dh] = acc + residual
    #pragma unroll
    for (int oi = 0; oi < 4; oi++) {
        int dh = sl * 64 + oi * 16 + l16;
        int col = h * Dh + dh;
        #pragma unroll
        for (int v = 0; v < 8; v++) {
            int row = it * 16 + half * 8 + v;
            size_t off = ((size_t)b * N_ + row) * D_ + col;
            out[off] = acc[oi][v] + res[off];
        }
    }
}

extern "C" void kernel_launch(void* const* d_in, const int* in_sizes, int n_in,
                              void* d_out, int out_size, void* d_ws, size_t ws_size,
                              hipStream_t stream) {
    (void)in_sizes; (void)n_in; (void)out_size; (void)ws_size;
    const float* x      = (const float*)d_in[0];
    const float* bias   = (const float*)d_in[1];
    const float* W1     = (const float*)d_in[2];
    const float* a_src1 = (const float*)d_in[3];
    const float* a_dst1 = (const float*)d_in[4];
    const float* g1     = (const float*)d_in[5];
    const float* b1     = (const float*)d_in[6];
    const float* W2     = (const float*)d_in[7];
    const float* a_src2 = (const float*)d_in[8];
    const float* a_dst2 = (const float*)d_in[9];
    const float* g2     = (const float*)d_in[10];
    const float* b2     = (const float*)d_in[11];
    float* outp = (float*)d_out;

    char* ws = (char*)d_ws;
    size_t off = 0;
    auto alloc = [&](size_t bytes) {
        void* p = ws + off;
        off += (bytes + 255) & ~(size_t)255;
        return p;
    };
    unsigned short* xnbf = (unsigned short*)alloc((size_t)B_ * N_ * D_ * 2);
    unsigned short* Wt   = (unsigned short*)alloc((size_t)D_ * D_ * 2);
    unsigned short* hbt  = (unsigned short*)alloc((size_t)B_ * D_ * N_ * 2);
    float* s     = (float*)alloc((size_t)B_ * 8 * N_ * 4);
    float* t     = (float*)alloc((size_t)B_ * 8 * N_ * 4);
    float* rm    = (float*)alloc((size_t)B_ * 8 * N_ * 4);
    float* rl    = (float*)alloc((size_t)B_ * 8 * N_ * 4);
    float* attn1 = (float*)alloc((size_t)B_ * N_ * D_ * 4);

    // ---- Layer 1 (H=8, Dh=64, 1 slice) ----
    ln_bf16_kernel<<<B_ * N_ / 8, 256, 0, stream>>>(x, g1, b1, xnbf);
    wtrans_kernel<<<D_ * D_ / 256, 256, 0, stream>>>(W1, Wt);
    gemm_kernel<<<B_ * (N_ / 128) * (D_ / 64), 256, 0, stream>>>(xnbf, Wt, hbt);
    st_kernel<<<B_ * 8 * N_ / 256, 256, 0, stream>>>(hbt, a_src1, a_dst1, s, t, 8, 64);
    stats_kernel<<<B_ * 8 * N_ / 8, 256, 0, stream>>>(s, t, bias, rm, rl, 8);
    pv_kernel<<<B_ * 8 * (N_ / 128) * 1, 256, 0, stream>>>(
        s, t, bias, rm, rl, hbt, x, attn1, 8, 64, 1);

    // ---- Layer 2 (H=1, Dh=512, 8 slices) ----
    ln_bf16_kernel<<<B_ * N_ / 8, 256, 0, stream>>>(attn1, g2, b2, xnbf);
    wtrans_kernel<<<D_ * D_ / 256, 256, 0, stream>>>(W2, Wt);
    gemm_kernel<<<B_ * (N_ / 128) * (D_ / 64), 256, 0, stream>>>(xnbf, Wt, hbt);
    st_kernel<<<B_ * 1 * N_ / 256, 256, 0, stream>>>(hbt, a_src2, a_dst2, s, t, 1, 512);
    stats_kernel<<<B_ * 1 * N_ / 8, 256, 0, stream>>>(s, t, bias, rm, rl, 1);
    pv_kernel<<<B_ * 1 * (N_ / 128) * 8, 256, 0, stream>>>(
        s, t, bias, rm, rl, hbt, attn1, outp, 1, 512, 8);
}
